// SlidingWindowAttention_45019847197305
// MI455X (gfx1250) — matrix operand order repaired and checked by reading, not compiled, not run on hardware
//
#include <hip/hip_runtime.h>

#define SEQD   8192
#define DMODEL 1024
#define NHEADS 16
#define DHEAD  64
#define WIN    512
#define NBLK   16   // SEQD / WIN

typedef __attribute__((ext_vector_type(16))) __bf16 v16bf;
typedef __attribute__((ext_vector_type(8)))  float  v8f;

union FragU { unsigned u[8]; v16bf v; };
union F4U   { float4 f; __bf16 h[8]; };

// ---------------------------------------------------------------------------
// CDNA5 async global->LDS copy (16 B per lane), tracked by ASYNCcnt.
// LDS destination is the low 32 bits of the generic shared-pointer address
// (aperture mapping).  Wait with s_wait_asynccnt.
// ---------------------------------------------------------------------------
__device__ inline void async_copy_b128(void* lds_dst, const void* gsrc) {
  const unsigned lds_off = (unsigned)(size_t)lds_dst;
  asm volatile("global_load_async_to_lds_b128 %0, %1, off"
               :
               : "v"(lds_off), "v"(gsrc)
               : "memory");
}

__device__ inline void wait_asynccnt0() {
  asm volatile("s_wait_asynccnt 0x0" ::: "memory");
}

// ---------------------------------------------------------------------------
// WMMA fragment builders (bf16, wave32), this code's contiguous k order k = 16h + s (same for A and B).
// A (16x32, MxK): lanes 0-15 -> M=lane, VGPR r: K=2r,2r+1;
//                 lanes 16-31 -> same M, K=16+2r,17+2r.
// B (32x16, KxN): lanes 0-15 -> N=lane, VGPR r: K=2r,2r+1;
//                 lanes 16-31 -> K=16+2r,17+2r.  (B staged n-major in LDS.)
// C/D (16x16 f32): VGPR r, lanes 0-15 -> M=r, N=lane; lanes 16-31 -> M=r+8.
// ---------------------------------------------------------------------------
__device__ inline v16bf load_a_frag(const __bf16* p, int m0, int stride,
                                    int koff, int lane) {
  const int L = lane & 15, hl = lane >> 4;
  const __bf16* row = p + (m0 + L) * stride + koff;
  FragU f;
#pragma unroll
  for (int r = 0; r < 8; ++r)
    f.u[r] = *(const unsigned*)(row + hl * 16 + 2 * r);
  return f.v;
}

__device__ inline v16bf load_b_frag(const __bf16* p, int n0, int stride,
                                    int koff, int lane) {
  const int L = lane & 15, hl = lane >> 4;
  const __bf16* row = p + (n0 + L) * stride + koff;
  FragU f;
#pragma unroll
  for (int r = 0; r < 8; ++r)
    f.u[r] = *(const unsigned*)(row + hl * 16 + 2 * r);
  return f.v;
}

__device__ inline v8f wmma_bf16(v16bf a, v16bf b, v8f c) {
  return __builtin_amdgcn_wmma_f32_16x16x32_bf16(false, a, false, b,
                                                 (short)0, c, false, false);
}

// ---------------------------------------------------------------------------
// fp32 -> bf16 cast (grid-stride)
// ---------------------------------------------------------------------------
__global__ void cast_bf16_k(const float* __restrict__ src,
                            __bf16* __restrict__ dst, int n) {
  int i = blockIdx.x * blockDim.x + threadIdx.x;
  int stride = gridDim.x * blockDim.x;
  for (; i < n; i += stride) dst[i] = (__bf16)src[i];
}

// ---------------------------------------------------------------------------
// GEMM: out[s, o] = sum_d X[s,d] * W[o,d] + bias[o]
// Each WG computes a 64-row block and the column PAIR (g*64.., g*64+512..)
// so RoPE (pairs d <-> d+512) is lane-local in the epilogue.
// Tiles staged with global_load_async_to_lds_b128 (ASYNCcnt path).
// ---------------------------------------------------------------------------
template <bool ROPE, bool OUTBF16>
__global__ __launch_bounds__(128) void gemm_rope_k(
    const __bf16* __restrict__ X,    // [SEQD, DMODEL] bf16
    const __bf16* __restrict__ Wt,   // [DMODEL, DMODEL] bf16, row-major [o][d]
    const float*  __restrict__ bias, // [DMODEL] fp32
    void* __restrict__ outp) {
  __shared__ __bf16 Als[64 * 32];  // [m][k]
  __shared__ __bf16 Blo[64 * 32];  // [n][k]
  __shared__ __bf16 Bhi[64 * 32];  // [n][k]

  const int rb = blockIdx.x;   // 64-row block
  const int g  = blockIdx.y;   // column pair group (0..7)
  const int tid = threadIdx.x;
  const int w = tid >> 5, lane = tid & 31;

  v8f acc_lo[4] = {};
  v8f acc_hi[4] = {};

  for (int k0 = 0; k0 < DMODEL; k0 += 32) {
    __syncthreads();  // previous iteration's LDS fully consumed
#pragma unroll
    for (int s = tid; s < 256; s += 128) {
      const int row = s >> 2, c4 = (s & 3) * 8;
      async_copy_b128(&Als[row * 32 + c4],
                      &X[(size_t)(rb * 64 + row) * DMODEL + k0 + c4]);
      async_copy_b128(&Blo[row * 32 + c4],
                      &Wt[(size_t)(g * 64 + row) * DMODEL + k0 + c4]);
      async_copy_b128(&Bhi[row * 32 + c4],
                      &Wt[(size_t)(g * 64 + 512 + row) * DMODEL + k0 + c4]);
    }

    if (k0 + 32 < DMODEL)  // lowers to global_prefetch_b8
      __builtin_prefetch(&X[(size_t)(rb * 64 + (tid & 63)) * DMODEL + k0 + 32],
                         0, 1);

    wait_asynccnt0();  // ASYNCcnt not tracked by compiler waits
    __syncthreads();

    const v16bf a = load_a_frag(Als, w * 16, 32, 0, lane);
#pragma unroll
    for (int nt = 0; nt < 4; ++nt) {
      const v16bf bl = load_b_frag(Blo, nt * 16, 32, 0, lane);
      acc_lo[nt] = wmma_bf16(a, bl, acc_lo[nt]);
      const v16bf bh = load_b_frag(Bhi, nt * 16, 32, 0, lane);
      acc_hi[nt] = wmma_bf16(a, bh, acc_hi[nt]);
    }
  }

  const int L = lane & 15, hl = lane >> 4;
#pragma unroll
  for (int nt = 0; nt < 4; ++nt) {
    const int d = g * 64 + nt * 16 + L;  // lo column (< 512)
    const float blo = bias[d];
    const float bhi = bias[d + 512];
#pragma unroll
    for (int r = 0; r < 8; ++r) {
      const int grow = rb * 64 + w * 16 + r + hl * 8;
      float xlo = acc_lo[nt][r] + blo;
      float xhi = acc_hi[nt][r] + bhi;
      if (ROPE) {
        // inv_freq[d] = 10000^(-d/512) = exp(-d * ln(1e4)/512)
        const float inv = __expf(-(float)d * 0.0179889460f);
        const float ang = (float)grow * inv;
        float sn, cs;
        __sincosf(ang, &sn, &cs);
        const float o1 = xlo * cs - xhi * sn;
        const float o2 = xlo * sn + xhi * cs;
        xlo = o1;
        xhi = o2;
      }
      const size_t off = (size_t)grow * DMODEL + d;
      if (OUTBF16) {
        ((__bf16*)outp)[off] = (__bf16)xlo;
        ((__bf16*)outp)[off + 512] = (__bf16)xhi;
      } else {
        ((float*)outp)[off] = xlo;
        ((float*)outp)[off + 512] = xhi;
      }
    }
  }
}

// ---------------------------------------------------------------------------
// Banded (sliding-window) attention, flash-style online softmax.
// grid = (8 query subtiles of 64 rows, 16 window blocks, 16 heads)
// Each wave owns 16 query rows. Context kv tiles j with band overlap:
// j in [st, st+8] (and j >= 8 for block 0, where prev block doesn't exist).
// Q / K tiles staged via async-to-LDS; V staged transposed (sync ds stores).
// ---------------------------------------------------------------------------
__global__ __launch_bounds__(128) void attn_swa_k(
    const __bf16* __restrict__ Q, const __bf16* __restrict__ K,
    const __bf16* __restrict__ V, __bf16* __restrict__ O) {
  __shared__ __bf16 Qls[64 * 64];      // [i][d]
  __shared__ __bf16 Kls[64 * 64];      // [kvt][d]  (= B n-major for Q.K^T)
  __shared__ __bf16 Vt[64 * 64];       // [d][kvt]  (= B n-major for P.V)
  __shared__ __bf16 Pls[4 * 16 * 64];  // per-wave P staging

  const int st = blockIdx.x;  // query subtile (64 rows) within window block
  const int nb = blockIdx.y;  // window block
  const int h  = blockIdx.z;  // head
  const int tid = threadIdx.x, w = tid >> 5, lane = tid & 31;
  const int L = lane & 15, hl = lane >> 4;
  const int hcol = h * DHEAD;

  // Stage Q tile once (64 x 64), async; waited before first use below.
  for (int s = tid; s < 512; s += 128) {
    const int t = s >> 3, c = (s & 7) * 8;
    async_copy_b128(&Qls[t * 64 + c],
                    &Q[(size_t)(nb * WIN + st * 64 + t) * DMODEL + hcol + c]);
  }

  float mrow[8], lrow[8];
#pragma unroll
  for (int r = 0; r < 8; ++r) {
    mrow[r] = -1e30f;
    lrow[r] = 0.f;
  }
  v8f accO[4] = {};

  const int jlo = (nb == 0) ? ((st > 8) ? st : 8) : st;
  const int jhi = (st + 8 < 15) ? (st + 8) : 15;
  __bf16* Pw = &Pls[w * 16 * 64];

  for (int j = jlo; j <= jhi; ++j) {
    __syncthreads();  // previous tile fully consumed
    const long kvbase = ((long)nb - 1) * WIN + (long)j * 64;  // >= 0 in range
    for (int s = tid; s < 512; s += 128) {
      const int t = s >> 3, c = (s & 7) * 8;
      const size_t gsrc = (size_t)(kvbase + t) * DMODEL + hcol + c;
      async_copy_b128(&Kls[t * 64 + c], &K[gsrc]);
      F4U vv;
      vv.f = *(const float4*)&V[gsrc];
#pragma unroll
      for (int e = 0; e < 8; ++e) Vt[(c + e) * 64 + t] = vv.h[e];
    }
    wait_asynccnt0();  // K (and first-iter Q) async copies complete
    __syncthreads();

    // S = Q . K^T   (K dim = Dh = 64 -> 2 WMMA k-steps)
    v8f sc[4] = {};
#pragma unroll
    for (int ks = 0; ks < 64; ks += 32) {
      const v16bf a = load_a_frag(Qls, w * 16, 64, ks, lane);
#pragma unroll
      for (int nt = 0; nt < 4; ++nt) {
        const v16bf b = load_b_frag(Kls, nt * 16, 64, ks, lane);
        sc[nt] = wmma_bf16(a, b, sc[nt]);
      }
    }

    // Band mask + online softmax. Row (r + hl*8) lives across one 16-lane half.
#pragma unroll
    for (int r = 0; r < 8; ++r) {
      const int i = st * 64 + w * 16 + r + hl * 8;  // query idx in block [0,512)
      float vals[4];
      float rmax = -1e30f;
#pragma unroll
      for (int nt = 0; nt < 4; ++nt) {
        const int jc = j * 64 + nt * 16 + L;  // context idx [0,1024)
        float x = sc[nt][r] * 0.125f;         // 1/sqrt(64)
        const bool ok = (jc >= i) && (jc <= i + WIN);
        x = ok ? x : -1e30f;
        vals[nt] = x;
        rmax = fmaxf(rmax, x);
      }
#pragma unroll
      for (int off = 8; off >= 1; off >>= 1)
        rmax = fmaxf(rmax, __shfl_xor(rmax, off, 16));
      const float mnew = fmaxf(mrow[r], rmax);
      const float alpha = __expf(mrow[r] - mnew);
      mrow[r] = mnew;
      float rsum = 0.f;
#pragma unroll
      for (int nt = 0; nt < 4; ++nt) {
        const float p = __expf(vals[nt] - mnew);
        rsum += p;
        Pw[(r + hl * 8) * 64 + nt * 16 + L] = (__bf16)p;
        accO[nt][r] *= alpha;
      }
#pragma unroll
      for (int off = 8; off >= 1; off >>= 1)
        rsum += __shfl_xor(rsum, off, 16);
      lrow[r] = lrow[r] * alpha + rsum;
    }

    // O += P . V   (K dim = 64 kv positions -> 2 WMMA k-steps)
#pragma unroll
    for (int ks = 0; ks < 64; ks += 32) {
      const v16bf a = load_a_frag(Pw, 0, 64, ks, lane);
#pragma unroll
      for (int nt = 0; nt < 4; ++nt) {
        const v16bf b = load_b_frag(Vt, nt * 16, 64, ks, lane);
        accO[nt] = wmma_bf16(a, b, accO[nt]);
      }
    }
  }

  // Normalize and write bf16 attention output in [S, DMODEL] layout
#pragma unroll
  for (int nt = 0; nt < 4; ++nt) {
#pragma unroll
    for (int r = 0; r < 8; ++r) {
      const int grow = nb * WIN + st * 64 + w * 16 + r + hl * 8;
      const int col = hcol + nt * 16 + L;
      const float o = accO[nt][r] / lrow[r];
      O[(size_t)grow * DMODEL + col] = (__bf16)o;
    }
  }
}

// ---------------------------------------------------------------------------
extern "C" void kernel_launch(void* const* d_in, const int* in_sizes, int n_in,
                              void* d_out, int out_size, void* d_ws,
                              size_t ws_size, hipStream_t stream) {
  (void)in_sizes; (void)n_in; (void)out_size; (void)ws_size;

  const float* hidden = (const float*)d_in[0];
  const float* Wq = (const float*)d_in[1];
  const float* bq = (const float*)d_in[2];
  const float* Wk = (const float*)d_in[3];
  const float* bk = (const float*)d_in[4];
  const float* Wv = (const float*)d_in[5];
  const float* bv = (const float*)d_in[6];
  const float* Wo = (const float*)d_in[7];
  const float* bo = (const float*)d_in[8];

  const size_t SD = (size_t)SEQD * DMODEL;    // 8M elems
  const size_t DD = (size_t)DMODEL * DMODEL;  // 1M elems
  __bf16* hb  = (__bf16*)d_ws;
  __bf16* wqb = hb + SD;
  __bf16* wkb = wqb + DD;
  __bf16* wvb = wkb + DD;
  __bf16* wob = wvb + DD;
  __bf16* qb  = wob + DD;
  __bf16* kb  = qb + SD;
  __bf16* vb  = kb + SD;
  __bf16* ab  = vb + SD;   // total ~88 MB of workspace

  cast_bf16_k<<<2048, 256, 0, stream>>>(hidden, hb, (int)SD);
  cast_bf16_k<<<1024, 256, 0, stream>>>(Wq, wqb, (int)DD);
  cast_bf16_k<<<1024, 256, 0, stream>>>(Wk, wkb, (int)DD);
  cast_bf16_k<<<1024, 256, 0, stream>>>(Wv, wvb, (int)DD);
  cast_bf16_k<<<1024, 256, 0, stream>>>(Wo, wob, (int)DD);

  const dim3 gg(SEQD / 64, 8);
  gemm_rope_k<true, true><<<gg, 128, 0, stream>>>(hb, wqb, bq, qb);
  gemm_rope_k<true, true><<<gg, 128, 0, stream>>>(hb, wkb, bk, kb);
  gemm_rope_k<false, true><<<gg, 128, 0, stream>>>(hb, wvb, bv, vb);

  attn_swa_k<<<dim3(8, NBLK, NHEADS), 128, 0, stream>>>(qb, kb, vb, ab);

  gemm_rope_k<false, false><<<gg, 128, 0, stream>>>(ab, wob, bo, d_out);
}
